// VectorQuantizer_26087631356134
// MI455X (gfx1250) — matrix operand order repaired and checked by reading, not compiled, not run on hardware
//
#include <hip/hip_runtime.h>
#include <hip/hip_bf16.h>

// Vector-quantizer forward for MI455X (gfx1250, wave32, WMMA + async-to-LDS).
// Pipeline: prep (W->f16 + ||w||^2)  ->  WMMA distance-GEMM fused argmin
//           ->  gather + deterministic deviation reduction -> final scalar.

typedef _Float16 v16h __attribute__((ext_vector_type(16)));
typedef _Float16 v8h  __attribute__((ext_vector_type(8)));
typedef _Float16 v4h  __attribute__((ext_vector_type(4)));
typedef float    v8f  __attribute__((ext_vector_type(8)));

#define FEATURES 256
#define SYMBOLS  1024
#define PIXELS   65536           // 16*64*64
#define NUMEL    16777216        // PIXELS*FEATURES
#define ROWS_PER_BLOCK 128       // 8 waves * 16 rows
#define GEMM_BLOCKS (PIXELS / ROWS_PER_BLOCK)   // 512
#define GATHER_BLOCKS (PIXELS / 8)              // 8192
#define NITER (SYMBOLS / 16)     // 64 symbol slabs
#define SLAB_HALVES (16 * FEATURES)             // 4096 f16 = 8 KB

// ---------------------------------------------------------------------------
// Kernel 1: W (f32) -> f16 copy + per-symbol squared norm (f32).
// ---------------------------------------------------------------------------
__global__ __launch_bounds__(256) void vq_prep(const float* __restrict__ W,
                                               _Float16* __restrict__ Wh,
                                               float* __restrict__ wn) {
  const int s = blockIdx.x;
  const int t = threadIdx.x;
  const float v = W[(size_t)s * FEATURES + t];
  Wh[(size_t)s * FEATURES + t] = (_Float16)v;
  float sq = v * v;
#pragma unroll
  for (int m = 1; m <= 16; m <<= 1) sq += __shfl_xor(sq, m, 32);
  __shared__ float red[8];
  if ((t & 31) == 0) red[t >> 5] = sq;
  __syncthreads();
  if (t == 0) {
    float a = 0.f;
#pragma unroll
    for (int i = 0; i < 8; ++i) a += red[i];
    wn[s] = a;
  }
}

// ---------------------------------------------------------------------------
// Kernel 2: distance GEMM (x . W^T via v_wmma_f32_16x16x32_f16) fused with
// argmin.  A tile hoisted into registers for the whole symbol loop; codebook
// slabs double-buffered in LDS via global_load_async_to_lds_b128 (ASYNCcnt).
// ---------------------------------------------------------------------------
__global__ __launch_bounds__(256) void vq_gemm_argmin(const float* __restrict__ x,
                                                      const _Float16* __restrict__ Wh,
                                                      const float* __restrict__ wn,
                                                      int* __restrict__ codes) {
  __shared__ _Float16 ldsA[8 * 16 * FEATURES];   // 64 KB: per-wave A staging
  __shared__ _Float16 ldsB[2 * SLAB_HALVES];     // 16 KB: double-buffered slab

  const int lane = threadIdx.x & 31;
  const int wave = threadIdx.x >> 5;
  const int rowBase = blockIdx.x * ROWS_PER_BLOCK + wave * 16;
  _Float16* myA = &ldsA[wave * 16 * FEATURES];

  // ---- Per-thread slab-staging geometry (256 threads cover 16x256 halves) --
  const int srow = threadIdx.x >> 4;             // symbol row in slab 0..15
  const int scol = (threadIdx.x & 15) * 16;      // 16 halves (32 B) per thread
  const unsigned ldsoff0 =
      (unsigned)(size_t)(&ldsB[srow * FEATURES + scol]);  // flat[31:0] = LDS addr
  const _Float16* gsrc0 = Wh + (size_t)srow * FEATURES + scol;

  // ---- Stage + convert this wave's 16x256 f32 x-tile into f16 LDS ----
  {
    const int r = lane >> 1;                 // row 0..15
    const int cbase = (lane & 1) * 128;      // half-row per lane
    const float* src = x + (size_t)(rowBase + r) * FEATURES + cbase;
    _Float16* dst = myA + r * FEATURES + cbase;
#pragma unroll
    for (int j = 0; j < 32; ++j) {
      const float4 v = ((const float4*)src)[j];
      v4h h;
      h[0] = (_Float16)v.x; h[1] = (_Float16)v.y;
      h[2] = (_Float16)v.z; h[3] = (_Float16)v.w;
      *(v4h*)(dst + j * 4) = h;
    }
  }

  // ---- Prefetch slab 0 (buffer 0) while we pack A fragments ----
  asm volatile("global_load_async_to_lds_b128 %0, %1, off\n\t"
               "global_load_async_to_lds_b128 %0, %1, off offset:16"
               :: "v"(ldsoff0), "v"(gsrc0) : "memory");

  const int ksel_a = (lane >= 16) ? 8 : 0;    // A frag K-offset (ISA 7.12.2)
  const int ksel_b = (lane >= 16) ? 8 : 0;    // B frag K-offset

  // ---- Hoist A fragments into registers (invariant over all symbols).
  // Same-wave DS ops are in-order, so these loads see our own ds_stores.
  v16h afrag[8];
#pragma unroll
  for (int kc = 0; kc < 8; ++kc) {
    const v8h* ap = (const v8h*)(myA + (lane & 15) * FEATURES + kc * 32 + ksel_a);
    const v8h alo = ap[0];
    const v8h ahi = ap[2];  // +16 halves
#pragma unroll
    for (int i = 0; i < 8; ++i) { afrag[kc][i] = alo[i]; afrag[kc][i + 8] = ahi[i]; }
  }

  float bestd[8];
  int   bestn[8];
#pragma unroll
  for (int r = 0; r < 8; ++r) { bestd[r] = 3.4e38f; bestn[r] = 0; }

  const int nlane = lane & 15;

  for (int it = 0; it < NITER; ++it) {
    // Our async issues have landed; barrier => everyone's landed AND the
    // other buffer (prefetch target) is no longer being read.
    asm volatile("s_wait_asynccnt 0x0" ::: "memory");
    __syncthreads();

    if (it + 1 < NITER) {   // prefetch next slab into the other buffer
      const unsigned nlds = ldsoff0 + (unsigned)(((it + 1) & 1) * SLAB_HALVES * 2);
      const _Float16* ng = gsrc0 + (size_t)(it + 1) * 16 * FEATURES;
      asm volatile("global_load_async_to_lds_b128 %0, %1, off\n\t"
                   "global_load_async_to_lds_b128 %0, %1, off offset:16"
                   :: "v"(nlds), "v"(ng) : "memory");
    }

    const _Float16* bbase =
        &ldsB[(it & 1) * SLAB_HALVES] + (lane & 15) * FEATURES + ksel_b;

    v8f acc = {};
#pragma unroll
    for (int kc = 0; kc < 8; ++kc) {
      // B fragment: lane holds symbol (lane&15), K = k0 + {0..7,16..23} / {8..15,24..31}.
      const v8h* bp = (const v8h*)(bbase + kc * 32);
      const v8h blo = bp[0];
      const v8h bhi = bp[2];  // +16 halves
      v16h b;
#pragma unroll
      for (int i = 0; i < 8; ++i) { b[i] = blo[i]; b[i + 8] = bhi[i]; }
      acc = __builtin_amdgcn_wmma_f32_16x16x32_f16(
          /*neg_a=*/false, afrag[kc], /*neg_b=*/false, b,
          /*c_mod=*/(short)0, acc, /*reuse_a=*/false, /*reuse_b=*/false);
    }

    const int s = it * 16 + nlane;
    const float w2 = wn[s];
#pragma unroll
    for (int r = 0; r < 8; ++r) {
      const float d = w2 - 2.0f * acc[r];   // ||w||^2 - 2 x.w (||x||^2 const)
      if (d < bestd[r]) { bestd[r] = d; bestn[r] = s; }  // ascending s keeps first
    }
  }

  // ---- Cross-lane min-reduce within each 16-lane half (N dimension) ----
#pragma unroll
  for (int mask = 1; mask <= 8; mask <<= 1) {
#pragma unroll
    for (int r = 0; r < 8; ++r) {
      const float od = __shfl_xor(bestd[r], mask, 32);
      const int   on = __shfl_xor(bestn[r], mask, 32);
      if (od < bestd[r] || (od == bestd[r] && on < bestn[r])) {
        bestd[r] = od; bestn[r] = on;
      }
    }
  }

  // D layout: VGPR r -> row r (lanes 0-15) / row r+8 (lanes 16-31).
  if ((lane & 15) == 0) {
    const int mbase = rowBase + ((lane >> 4) ? 8 : 0);
#pragma unroll
    for (int r = 0; r < 8; ++r) codes[mbase + r] = bestn[r];
  }
}

// ---------------------------------------------------------------------------
// Kernel 3: out[p,:] = W[codes[p],:] (f32) + exact per-block sum of (x-w)^2.
// One wave per pixel; deterministic shfl + LDS reduction -> partials[block].
// ---------------------------------------------------------------------------
__global__ __launch_bounds__(256) void vq_gather_dev(const float* __restrict__ x,
                                                     const float* __restrict__ W,
                                                     const int* __restrict__ codes,
                                                     float* __restrict__ out,
                                                     float* __restrict__ partials) {
  const int lane = threadIdx.x & 31;
  const int wave = threadIdx.x >> 5;
  const int p = blockIdx.x * 8 + wave;
  const int code = codes[p];
  const float4* wp = (const float4*)(W + (size_t)code * FEATURES);
  const float4* xp = (const float4*)(x + (size_t)p * FEATURES);
  float4* op = (float4*)(out + (size_t)p * FEATURES);
  float acc = 0.f;
#pragma unroll
  for (int j = 0; j < 2; ++j) {
    const int idx = lane * 2 + j;           // 64 float4 per row / 32 lanes
    const float4 wv = wp[idx];
    const float4 xv = xp[idx];
    op[idx] = wv;
    const float d0 = xv.x - wv.x, d1 = xv.y - wv.y;
    const float d2 = xv.z - wv.z, d3 = xv.w - wv.w;
    acc += d0 * d0 + d1 * d1 + d2 * d2 + d3 * d3;
  }
#pragma unroll
  for (int m = 1; m <= 16; m <<= 1) acc += __shfl_xor(acc, m, 32);
  __shared__ float red[8];
  if (lane == 0) red[wave] = acc;
  __syncthreads();
  if (threadIdx.x == 0) {
    float s = 0.f;
#pragma unroll
    for (int i = 0; i < 8; ++i) s += red[i];
    partials[blockIdx.x] = s;
  }
}

// ---------------------------------------------------------------------------
// Kernel 4: deterministic reduction of 8192 partials -> deviation scalar.
// ---------------------------------------------------------------------------
__global__ __launch_bounds__(256) void vq_final(const float* __restrict__ partials,
                                                float* __restrict__ devout) {
  __shared__ float red[256];
  float s = 0.f;
  for (int i = threadIdx.x; i < GATHER_BLOCKS; i += 256) s += partials[i];
  red[threadIdx.x] = s;
  __syncthreads();
  for (int k = 128; k > 0; k >>= 1) {
    if (threadIdx.x < k) red[threadIdx.x] += red[threadIdx.x + k];
    __syncthreads();
  }
  if (threadIdx.x == 0) devout[0] = red[0] * (1.25f / (float)NUMEL);
}

// ---------------------------------------------------------------------------
extern "C" void kernel_launch(void* const* d_in, const int* in_sizes, int n_in,
                              void* d_out, int out_size, void* d_ws, size_t ws_size,
                              hipStream_t stream) {
  (void)in_sizes; (void)n_in; (void)out_size; (void)ws_size;
  const float* x = (const float*)d_in[0];   // [65536, 256] f32
  const float* W = (const float*)d_in[1];   // [1024, 256]  f32
  float* out = (float*)d_out;               // [65536*256] + 1 (deviation)

  char* ws = (char*)d_ws;
  _Float16* Wh    = (_Float16*)ws;                  // 512 KB f16 codebook
  float*    wn    = (float*)(ws + 524288);          //   4 KB ||w||^2
  int*      codes = (int*)(ws + 528384);            // 256 KB argmin codes
  float*    parts = (float*)(ws + 790528);          //  32 KB block partials

  vq_prep<<<SYMBOLS, 256, 0, stream>>>(W, Wh, wn);
  vq_gemm_argmin<<<GEMM_BLOCKS, 256, 0, stream>>>(x, Wh, wn, codes);
  vq_gather_dev<<<GATHER_BLOCKS, 256, 0, stream>>>(x, W, codes, out, parts);
  vq_final<<<1, 256, 0, stream>>>(parts, out + (size_t)NUMEL);
}
